// StateSpaceByteCore_7816840479153
// MI455X (gfx1250) — compile-verified, matrix operand order checked by reading, not run on hardware
//
#include <hip/hip_runtime.h>

// ---------------------------------------------------------------------------
// Types for CDNA5 WMMA (gfx1250, wave32)
// ---------------------------------------------------------------------------
typedef __bf16 bf16_t;
typedef bf16_t v16bf __attribute__((ext_vector_type(16)));
typedef bf16_t v8bf  __attribute__((ext_vector_type(8)));
typedef bf16_t v4bf  __attribute__((ext_vector_type(4)));
typedef float  v8f   __attribute__((ext_vector_type(8)));
typedef int    v4i_t __attribute__((vector_size(16)));   // matches async builtin param

#define AS_GLOBAL __attribute__((address_space(1)))
#define AS_LDS    __attribute__((address_space(3)))

#define D_C    1024            // model dim (K and N)
#define D_T    4096
#define D_B    8
#define D_M    (D_B * D_T)     // 32768 rows of X
#define TILE_M 128
#define TILE_NH 64             // N tile, applied to BOTH weight matrices
#define TILE_K 32
#define PADK   40              // 32 + 8 bf16 pad -> 80B row stride (16B aligned)

// CDNA5 async global->LDS path (ASYNCcnt-tracked, no VGPR round trip)
#if __has_builtin(__builtin_amdgcn_global_load_async_to_lds_b128) && \
    __has_builtin(__builtin_amdgcn_s_wait_asynccnt)
#define HAVE_ASYNC_LDS 1
#else
#define HAVE_ASYNC_LDS 0
#endif

__device__ __forceinline__ v8f wmma_bf16(v16bf a, v16bf b, v8f c) {
  return __builtin_amdgcn_wmma_f32_16x16x32_bf16(
      /*neg_a=*/false, a, /*neg_b=*/false, b,
      /*c_mod=*/(short)0, c, /*reuse_a=*/false, /*reuse_b=*/false);
}

// ---------------------------------------------------------------------------
// Kernel 1: split fp32 -> bf16 hi + bf16 lo (hi = rne(x), lo = rne(x - hi))
// ---------------------------------------------------------------------------
__global__ void split_bf16_kernel(const float* __restrict__ in,
                                  bf16_t* __restrict__ hi,
                                  bf16_t* __restrict__ lo,
                                  long n4) {
  long i = (long)blockIdx.x * blockDim.x + threadIdx.x;
  if (i >= n4) return;
  const float4 v = reinterpret_cast<const float4*>(in)[i];
  float f[4] = {v.x, v.y, v.z, v.w};
  v4bf h, l;
#pragma unroll
  for (int k = 0; k < 4; ++k) {
    bf16_t hh = (bf16_t)f[k];
    h[k] = hh;
    l[k] = (bf16_t)(f[k] - (float)hh);
  }
  reinterpret_cast<v4bf*>(hi)[i] = h;
  reinterpret_cast<v4bf*>(lo)[i] = l;
}

// ---------------------------------------------------------------------------
// GEMM helpers
// ---------------------------------------------------------------------------
__device__ __forceinline__ void cp16(bf16_t* sdst, const bf16_t* gsrc) {
#if HAVE_ASYNC_LDS
  // global_load_async_to_lds_b128: memory -> LDS, tracked with ASYNCcnt
  __builtin_amdgcn_global_load_async_to_lds_b128(
      (AS_GLOBAL v4i_t*)gsrc, (AS_LDS v4i_t*)sdst, 0, 0);
#else
  *reinterpret_cast<float4*>(sdst) = *reinterpret_cast<const float4*>(gsrc);
#endif
}

// Stage one K-slab of all six tiles (A hi/lo 128x32, 4x B tiles 64x32).
// 256 threads x 8 async b128 copies each.
__device__ __forceinline__ void stage(
    int tid, int mBlock, int nBlock, int kb,
    const bf16_t* __restrict__ Ah, const bf16_t* __restrict__ Al,
    const bf16_t* __restrict__ Sh, const bf16_t* __restrict__ Sl,
    const bf16_t* __restrict__ Gh, const bf16_t* __restrict__ Gl,
    bf16_t* sAh, bf16_t* sAl, bf16_t* sSh, bf16_t* sSl, bf16_t* sGh, bf16_t* sGl) {
  const bf16_t* gs[8];
  bf16_t*       sd[8];
  {
    // A tiles: 128 rows x 4 chunks = 512 chunks; this thread takes tid, tid+256
    const int c0 = tid, c1 = tid + 256;
    const int r0 = c0 >> 2, k0 = (c0 & 3) << 3;
    const int r1 = c1 >> 2, k1 = (c1 & 3) << 3;
    const size_t g0 = (size_t)(mBlock + r0) * D_C + kb + k0;
    const size_t g1 = (size_t)(mBlock + r1) * D_C + kb + k1;
    gs[0] = Ah + g0; sd[0] = sAh + r0 * PADK + k0;
    gs[1] = Ah + g1; sd[1] = sAh + r1 * PADK + k1;
    gs[2] = Al + g0; sd[2] = sAl + r0 * PADK + k0;
    gs[3] = Al + g1; sd[3] = sAl + r1 * PADK + k1;
  }
  {
    // B tiles: 64 rows x 4 chunks = 256 chunks; this thread takes chunk tid
    const int r = tid >> 2, kc = (tid & 3) << 3;
    const size_t go = (size_t)(nBlock + r) * D_C + kb + kc;
    const int    so = r * PADK + kc;
    gs[4] = Sh + go; sd[4] = sSh + so;
    gs[5] = Sl + go; sd[5] = sSl + so;
    gs[6] = Gh + go; sd[6] = sGh + so;
    gs[7] = Gl + go; sd[7] = sGl + so;
  }
#if HAVE_ASYNC_LDS
#pragma unroll
  for (int i = 0; i < 8; ++i) cp16(sd[i], gs[i]);
#else
  float4 t[8];
#pragma unroll
  for (int i = 0; i < 8; ++i) t[i] = *reinterpret_cast<const float4*>(gs[i]);
#pragma unroll
  for (int i = 0; i < 8; ++i) *reinterpret_cast<float4*>(sd[i]) = t[i];
#endif
}

__device__ __forceinline__ void wait_stage() {
#if HAVE_ASYNC_LDS
  __builtin_amdgcn_s_wait_asynccnt(0);
#endif
  __syncthreads();
}

// LDS -> fragment, matching the ISA 16-bit A layout:
// lanes 0-15 hold K {0..7, 16..23}; lanes 16-31 hold K {8..15, 24..31}.
__device__ __forceinline__ v16bf ld_frag(const bf16_t* s, int row0, int lane) {
  const bf16_t* p = s + (row0 + (lane & 15)) * PADK + ((lane & 16) ? 8 : 0);
  v8bf a = *reinterpret_cast<const v8bf*>(p);
  v8bf b = *reinterpret_cast<const v8bf*>(p + 16);
  return __builtin_shufflevector(a, b, 0, 1, 2, 3, 4, 5, 6, 7,
                                       8, 9, 10, 11, 12, 13, 14, 15);
}

// ---------------------------------------------------------------------------
// Kernel 2: fused dual split-precision bf16x3 GEMM.
//   px = X @ Ws^T + bs ; g = sigmoid(X @ Wg^T + bg)
//   Block tile: M=128 x N=64, covering BOTH weight matrices (X staged once).
//   8 waves in 4(M) x 2(N); wave tile 32x32 per matrix; double-buffered LDS
//   fed by GLOBAL_LOAD_ASYNC_TO_LDS_B128 (ASYNCcnt pipeline).
// ---------------------------------------------------------------------------
__global__ void __launch_bounds__(256)
fused_dual_gemm(const bf16_t* __restrict__ Ah, const bf16_t* __restrict__ Al,
                const bf16_t* __restrict__ Sh, const bf16_t* __restrict__ Sl,
                const bf16_t* __restrict__ Gh, const bf16_t* __restrict__ Gl,
                const float* __restrict__ bs, const float* __restrict__ bg,
                float* __restrict__ px, float* __restrict__ g) {
  __shared__ __align__(16) bf16_t sAh[2][TILE_M * PADK];
  __shared__ __align__(16) bf16_t sAl[2][TILE_M * PADK];
  __shared__ __align__(16) bf16_t sSh[2][TILE_NH * PADK];
  __shared__ __align__(16) bf16_t sSl[2][TILE_NH * PADK];
  __shared__ __align__(16) bf16_t sGh[2][TILE_NH * PADK];
  __shared__ __align__(16) bf16_t sGl[2][TILE_NH * PADK];

  const int tid    = threadIdx.x;
  const int lane   = tid & 31;          // wave32
  const int wid    = tid >> 5;          // 0..7
  const int wm     = wid >> 1;          // 0..3 -> 32 rows
  const int wn     = wid & 1;           // 0..1 -> 32 cols
  const int mBlock = blockIdx.y * TILE_M;
  const int nBlock = blockIdx.x * TILE_NH;

  v8f accS[2][2] = {};
  v8f accG[2][2] = {};

  // Prologue: stage K-slab 0 into buffer 0
  stage(tid, mBlock, nBlock, 0, Ah, Al, Sh, Sl, Gh, Gl,
        sAh[0], sAl[0], sSh[0], sSl[0], sGh[0], sGl[0]);
  wait_stage();

  int cur = 0;
  for (int kb = 0; kb < D_C; kb += TILE_K) {
    const int nxt = cur ^ 1;
    if (kb + TILE_K < D_C) {
      // Overlap: async-copy the next K-slab while WMMAs run on the current one
      stage(tid, mBlock, nBlock, kb + TILE_K, Ah, Al, Sh, Sl, Gh, Gl,
            sAh[nxt], sAl[nxt], sSh[nxt], sSl[nxt], sGh[nxt], sGl[nxt]);
    }

    v16bf fah[2], fal[2];
#pragma unroll
    for (int i = 0; i < 2; ++i) {
      fah[i] = ld_frag(sAh[cur], wm * 32 + i * 16, lane);
      fal[i] = ld_frag(sAl[cur], wm * 32 + i * 16, lane);
    }
#pragma unroll
    for (int j = 0; j < 2; ++j) {
      const int r = wn * 32 + j * 16;
      const v16bf bsh = ld_frag(sSh[cur], r, lane);
      const v16bf bsl = ld_frag(sSl[cur], r, lane);
      const v16bf bgh = ld_frag(sGh[cur], r, lane);
      const v16bf bgl = ld_frag(sGl[cur], r, lane);
#pragma unroll
      for (int i = 0; i < 2; ++i) {
        accS[i][j] = wmma_bf16(fah[i], bsh, accS[i][j]);  // hi*hi
        accS[i][j] = wmma_bf16(fah[i], bsl, accS[i][j]);  // hi*lo
        accS[i][j] = wmma_bf16(fal[i], bsh, accS[i][j]);  // lo*hi
        accG[i][j] = wmma_bf16(fah[i], bgh, accG[i][j]);
        accG[i][j] = wmma_bf16(fah[i], bgl, accG[i][j]);
        accG[i][j] = wmma_bf16(fal[i], bgh, accG[i][j]);
      }
    }

    wait_stage();   // next slab landed in LDS; all waves done reading cur
    cur = nxt;
  }

  // Epilogue. C/D layout: lanes 0-15: M=r, N=lane; lanes 16-31: M=8+r, N=lane-16
#pragma unroll
  for (int i = 0; i < 2; ++i) {
    const int mrow = mBlock + wm * 32 + i * 16 + ((lane & 16) ? 8 : 0);
#pragma unroll
    for (int j = 0; j < 2; ++j) {
      const int ncol = nBlock + wn * 32 + j * 16 + (lane & 15);
      const float bsv = bs[ncol];
      const float bgv = bg[ncol];
#pragma unroll
      for (int r = 0; r < 8; ++r) {
        px[(size_t)(mrow + r) * D_C + ncol] = accS[i][j][r] + bsv;
        const float v = accG[i][j][r] + bgv;
        g[(size_t)(mrow + r) * D_C + ncol] = 1.0f / (1.0f + __expf(-v));
      }
    }
  }
}

// ---------------------------------------------------------------------------
// Kernel 3: sequential scan over T, parallel over B*C = 8192 lanes.
//   state = tanh(p_t + state); out_t = g_t * state
// ---------------------------------------------------------------------------
__global__ void scan_kernel(const float* __restrict__ px,
                            const float* __restrict__ g,
                            float* __restrict__ outs,
                            float* __restrict__ finals) {
  const int idx = blockIdx.x * blockDim.x + threadIdx.x;  // 0..8191
  const int b = idx >> 10;
  const int c = idx & (D_C - 1);
  const size_t base = (size_t)b * D_T * D_C + c;
  const float* pp = px + base;
  const float* pg = g + base;
  float* po = outs + base;

  float state = 0.0f;
  float p  = pp[0];
  float gg = pg[0];
  for (int t = 0; t < D_T; ++t) {
    float pn = 0.0f, gn = 0.0f;
    if (t + 1 < D_T) {           // uniform; loads issue before the tanh use
      pn = pp[(size_t)(t + 1) * D_C];
      gn = pg[(size_t)(t + 1) * D_C];
    }
    state = tanhf(p + state);
    po[(size_t)t * D_C] = gg * state;
    p = pn;
    gg = gn;
  }
  finals[idx] = state;
}

// ---------------------------------------------------------------------------
// Launch
// ---------------------------------------------------------------------------
extern "C" void kernel_launch(void* const* d_in, const int* in_sizes, int n_in,
                              void* d_out, int out_size, void* d_ws, size_t ws_size,
                              hipStream_t stream) {
  const float* x  = (const float*)d_in[0];
  const float* Ws = (const float*)d_in[1];
  const float* bs = (const float*)d_in[2];
  const float* Wg = (const float*)d_in[3];
  const float* bg = (const float*)d_in[4];

  const size_t nX = (size_t)D_M * D_C;   // 33,554,432 floats
  const size_t nW = (size_t)D_C * D_C;   //  1,048,576 floats

  char* ws = (char*)d_ws;
  float*  px   = (float*)(ws);                                 // 128 MiB
  float*  g    = (float*)(ws + nX * 4);                        // 128 MiB
  bf16_t* xhi  = (bf16_t*)(ws + nX * 8);                       //  64 MiB
  bf16_t* xlo  = (bf16_t*)(ws + nX * 8 + nX * 2);              //  64 MiB
  bf16_t* wshi = (bf16_t*)(ws + nX * 12);                      //   2 MiB x4
  bf16_t* wslo = (bf16_t*)(ws + nX * 12 + nW * 2);
  bf16_t* wghi = (bf16_t*)(ws + nX * 12 + nW * 4);
  bf16_t* wglo = (bf16_t*)(ws + nX * 12 + nW * 6);

  // 1) fp32 -> bf16 hi/lo splits
  {
    const long n4x = (long)(nX / 4);
    const long n4w = (long)(nW / 4);
    split_bf16_kernel<<<(n4x + 255) / 256, 256, 0, stream>>>(x, xhi, xlo, n4x);
    split_bf16_kernel<<<(n4w + 255) / 256, 256, 0, stream>>>(Ws, wshi, wslo, n4w);
    split_bf16_kernel<<<(n4w + 255) / 256, 256, 0, stream>>>(Wg, wghi, wglo, n4w);
  }

  // 2) fused dual GEMM: both projections from one X staging pass
  {
    dim3 grid(D_C / TILE_NH, D_M / TILE_M);  // (16, 256)
    fused_dual_gemm<<<grid, 256, 0, stream>>>(xhi, xlo, wshi, wslo, wghi, wglo,
                                              bs, bg, px, g);
  }

  // 3) recurrence scan -> outs [B,T,C] then final state [B,C]
  {
    float* outs   = (float*)d_out;
    float* finals = outs + (size_t)D_M * D_C;
    scan_kernel<<<256, 32, 0, stream>>>(px, g, outs, finals);  // one wave/block
  }
}
